// NaiveGNN_34961033790066
// MI455X (gfx1250) — matrix operand order repaired and checked by reading, not compiled, not run on hardware
//
#include <hip/hip_runtime.h>
#include <hip/hip_bf16.h>

typedef __attribute__((ext_vector_type(16))) _Float16 v16h;
typedef __attribute__((ext_vector_type(8)))  _Float16 v8h;
typedef __attribute__((ext_vector_type(8)))  float    v8f;

#define LN2F 0.69314718055994530942f

__device__ __forceinline__ v8f wmma16(v16h a, v16h b, v8f c) {
  // D = A(16x32 f16) x B(32x16 f16) + C(16x16 f32)
  return __builtin_amdgcn_wmma_f32_16x16x32_f16(false, a, false, b, (short)0, c, false, false);
}

__device__ __forceinline__ v8f v8zero() {
  v8f c;
#pragma unroll
  for (int i = 0; i < 8; ++i) c[i] = 0.f;
  return c;
}

// Branchless fast transcendentals on the v_exp/v_log/v_rcp trans pipes.
__device__ __forceinline__ float tanh_fast(float x) {
  float t = __expf(2.0f * x);                      // inf for large x, 0 for very neg
  return 1.0f - 2.0f * __builtin_amdgcn_rcpf(t + 1.0f);
}
// ShiftedSoftplus: softplus(x) - ln2
__device__ __forceinline__ float ssp_fast(float x) {
  return fmaxf(x, 0.f) + __logf(1.0f + __expf(-fabsf(x))) - LN2F;
}

// A-fragment (16x32 f16) from LDS row-major tile [16 x strideK].
// ISA layout = two contiguous 8-half runs: K=k0+8h+0..7 and K=k0+16+8h+0..7.
__device__ __forceinline__ v16h ldsA2(const _Float16* base, int strideK, int m, int h, int k0) {
  const _Float16* p = base + m * strideK + k0 + 8 * h;
  v8h lo = *(const v8h*)(p);
  v8h hi = *(const v8h*)(p + 16);
  v16h a;
#pragma unroll
  for (int e = 0; e < 8; ++e) { a[e] = lo[e]; a[8 + e] = hi[e]; }
  return a;
}

// B-fragment (32x16 f16) from LDS *transposed* weights Wt[n][k] (stride strideK).
// Lane needs K = k0+8h+0..7 and K = k0+16+8h+0..7 -> two 8-half runs of row n.
__device__ __forceinline__ v16h ldsBT(const _Float16* baseT, int strideK, int n, int h, int k0) {
  const _Float16* p = baseT + n * strideK + k0 + 8 * h;
  v8h lo = *(const v8h*)(p);
  v8h hi = *(const v8h*)(p + 16);
  v16h b;
#pragma unroll
  for (int e = 0; e < 8; ++e) { b[e] = lo[e]; b[8 + e] = hi[e]; }
  return b;
}

// A-fragment from a global f32 row (64 wide), cvt to f16 on the fly.
__device__ __forceinline__ v16h gA8(const float* grow, int h, int k0) {
  const float* plo = grow + k0 + 8 * h;
  const float* phi = grow + k0 + 16 + 8 * h;
  float4 l0 = *(const float4*)(plo);
  float4 l1 = *(const float4*)(plo + 4);
  float4 h0 = *(const float4*)(phi);
  float4 h1 = *(const float4*)(phi + 4);
  v16h a;
  a[0] = (_Float16)l0.x;  a[1] = (_Float16)l0.y;  a[2] = (_Float16)l0.z;  a[3] = (_Float16)l0.w;
  a[4] = (_Float16)l1.x;  a[5] = (_Float16)l1.y;  a[6] = (_Float16)l1.z;  a[7] = (_Float16)l1.w;
  a[8] = (_Float16)h0.x;  a[9] = (_Float16)h0.y;  a[10] = (_Float16)h0.z; a[11] = (_Float16)h0.w;
  a[12] = (_Float16)h1.x; a[13] = (_Float16)h1.y; a[14] = (_Float16)h1.z; a[15] = (_Float16)h1.w;
  return a;
}

// ---------------------------------------------------------------------------
// K1: hv = tanh(net_feat @ W_net + b_net) @ W_cfn + b_cfn      [N_NET x 64] f16
// ---------------------------------------------------------------------------
__global__ __launch_bounds__(128) void k_hv(
    const float* __restrict__ net_feat,
    const float* __restrict__ W_net, const float* __restrict__ b_net,
    const float* __restrict__ W_cfn, const float* __restrict__ b_cfn,
    _Float16* __restrict__ hv, int n_net) {
  __shared__ _Float16 sWnetT[64 * 32];   // transposed, K padded 4->32 with zeros
  __shared__ _Float16 sWcfnT[64 * 64];   // transposed [n][k]
  __shared__ float sBnet[64];
  __shared__ float sBcfn[64];
  __shared__ _Float16 sStage[4][16 * 64];

  const int tid = threadIdx.x;
  for (int i = tid; i < 64 * 32; i += 128) {
    int n = i >> 5, k = i & 31;
    sWnetT[i] = (k < 4) ? (_Float16)W_net[k * 64 + n] : (_Float16)0.f;
  }
  for (int i = tid; i < 64 * 64; i += 128) {
    int n = i >> 6, k = i & 63;
    sWcfnT[i] = (_Float16)W_cfn[k * 64 + n];
  }
  if (tid < 64) { sBnet[tid] = b_net[tid]; sBcfn[tid] = b_cfn[tid]; }
  __syncthreads();

  const int wave = tid >> 5, lane = tid & 31;
  const int mn = lane & 15, h = lane >> 4;
  const int tile = blockIdx.x * 4 + wave;
  const int row0 = tile * 16;
  if (row0 >= n_net) return;

  // GEMM1: A = net_feat rows (K=4, zero padded), one float4 per owning lane
  v16h a4;
#pragma unroll
  for (int e = 0; e < 16; ++e) a4[e] = (_Float16)0.f;
  if (h == 0) {
    int r = row0 + mn; r = r < n_net ? r : n_net - 1;
    float4 f = *(const float4*)(net_feat + (size_t)r * 4);
    a4[0] = (_Float16)f.x; a4[1] = (_Float16)f.y;
    a4[2] = (_Float16)f.z; a4[3] = (_Float16)f.w;
  }

  _Float16* stg = sStage[wave];
#pragma unroll
  for (int nt = 0; nt < 4; ++nt) {
    v8f c = wmma16(a4, ldsBT(sWnetT, 32, nt * 16 + mn, h, 0), v8zero());
    float bb = sBnet[nt * 16 + mn];
#pragma unroll
    for (int v = 0; v < 8; ++v)
      stg[(8 * h + v) * 64 + nt * 16 + mn] = (_Float16)tanh_fast(c[v] + bb);
  }
  asm volatile("s_wait_dscnt 0" ::: "memory");

  // GEMM2: K=64
  v16h a0 = ldsA2(stg, 64, mn, h, 0);
  v16h a1 = ldsA2(stg, 64, mn, h, 32);
#pragma unroll
  for (int nt = 0; nt < 4; ++nt) {
    v8f c = wmma16(a0, ldsBT(sWcfnT, 64, nt * 16 + mn, h, 0), v8zero());
    c = wmma16(a1, ldsBT(sWcfnT, 64, nt * 16 + mn, h, 32), c);
    float bb = sBcfn[nt * 16 + mn];
#pragma unroll
    for (int v = 0; v < 8; ++v) {
      int r = row0 + 8 * h + v;
      if (r < n_net) hv[(size_t)r * 64 + nt * 16 + mn] = (_Float16)(c[v] + bb);
    }
  }
}

// ---------------------------------------------------------------------------
// K2: fused CFConv edge pipeline over 'pinned' edges (16 edges per wave):
//   hp = tanh(pin_feat@W_pin+b_pin); t = ssp(hp@W_cfe1+b1); he = ssp(t@W_cfe2+b2)
//   atomicAdd(acc[pinned_dst], he * hv[pinned_src])
// ---------------------------------------------------------------------------
__global__ __launch_bounds__(128) void k_cfconv(
    const float* __restrict__ pin_feat,
    const int* __restrict__ esrc, const int* __restrict__ edst,
    const float* __restrict__ W_pin, const float* __restrict__ b_pin,
    const float* __restrict__ W_cfe1, const float* __restrict__ b_cfe1,
    const float* __restrict__ W_cfe2, const float* __restrict__ b_cfe2,
    const _Float16* __restrict__ hv, float* __restrict__ acc, int e_pin) {
  __shared__ float sWpin[4 * 16];
  __shared__ float sBpin[16];
  __shared__ _Float16 sW1T[64 * 32];   // transposed, K padded 16->32 with zeros
  __shared__ float sB1[64];
  __shared__ _Float16 sW2T[64 * 64];   // transposed [n][k]
  __shared__ float sB2[64];
  __shared__ _Float16 sHp[4][16 * 32]; // h_pin tile, K padded 16->32 with zeros
  __shared__ _Float16 sT[4][16 * 64];
  __shared__ int sS[4][16];
  __shared__ int sD[4][16];

  const int tid = threadIdx.x;
  for (int i = tid; i < 64; i += 128) sWpin[i] = W_pin[i];
  if (tid < 16) sBpin[tid] = b_pin[tid];
  for (int i = tid; i < 64 * 32; i += 128) {
    int n = i >> 5, k = i & 31;
    sW1T[i] = (k < 16) ? (_Float16)W_cfe1[k * 64 + n] : (_Float16)0.f;
  }
  for (int i = tid; i < 64 * 64; i += 128) {
    int n = i >> 6, k = i & 63;
    sW2T[i] = (_Float16)W_cfe2[k * 64 + n];
  }
  if (tid < 64) { sB1[tid] = b_cfe1[tid]; sB2[tid] = b_cfe2[tid]; }
  __syncthreads();

  const int wave = tid >> 5, lane = tid & 31;
  const int mn = lane & 15, h = lane >> 4;
  const long long tile = (long long)blockIdx.x * 4 + wave;
  const long long e0 = tile * 16;
  if (e0 >= e_pin) return;

  // Step 1: h_pin tile (16 edges x 16 feats) + zero pad + edge index stage
  {
    long long e = e0 + mn;
    long long ec = e < e_pin ? e : (long long)e_pin - 1;
    float4 x = *(const float4*)(pin_feat + ec * 4);
    int jb = 8 * h;
    v8h hp;
#pragma unroll
    for (int jj = 0; jj < 8; ++jj) {
      int j = jb + jj;
      float s = sBpin[j] + x.x * sWpin[j] + x.y * sWpin[16 + j]
              + x.z * sWpin[32 + j] + x.w * sWpin[48 + j];
      hp[jj] = (_Float16)tanh_fast(s);
    }
    *(v8h*)(&sHp[wave][mn * 32 + jb]) = hp;
    v8h zz;
#pragma unroll
    for (int jj = 0; jj < 8; ++jj) zz[jj] = (_Float16)0.f;
    *(v8h*)(&sHp[wave][mn * 32 + 16 + jb]) = zz;
    if (h == 0) { sS[wave][mn] = esrc[ec]; sD[wave][mn] = edst[ec]; }
  }
  asm volatile("s_wait_dscnt 0" ::: "memory");

  // Step 2: t = ssp(hp @ W_cfe1 + b1), K=16 (zero-padded to 32)
  {
    v16h a = ldsA2(sHp[wave], 32, mn, h, 0);
#pragma unroll
    for (int nt = 0; nt < 4; ++nt) {
      v8f c = wmma16(a, ldsBT(sW1T, 32, nt * 16 + mn, h, 0), v8zero());
      float bb = sB1[nt * 16 + mn];
#pragma unroll
      for (int v = 0; v < 8; ++v)
        sT[wave][(8 * h + v) * 64 + nt * 16 + mn] = (_Float16)ssp_fast(c[v] + bb);
    }
  }
  asm volatile("s_wait_dscnt 0" ::: "memory");

  // Step 3: he = ssp(t @ W_cfe2 + b2), K=64
  v8f he[4];
  {
    v16h a0 = ldsA2(sT[wave], 64, mn, h, 0);
    v16h a1 = ldsA2(sT[wave], 64, mn, h, 32);
#pragma unroll
    for (int nt = 0; nt < 4; ++nt) {
      v8f c = wmma16(a0, ldsBT(sW2T, 64, nt * 16 + mn, h, 0), v8zero());
      c = wmma16(a1, ldsBT(sW2T, 64, nt * 16 + mn, h, 32), c);
      float bb = sB2[nt * 16 + mn];
#pragma unroll
      for (int v = 0; v < 8; ++v) he[nt][v] = ssp_fast(c[v] + bb);
    }
  }

  // Step 4: msg = he * hv[src], L2-resident atomic scatter-add into acc[dst]
#pragma unroll
  for (int v = 0; v < 8; ++v) {
    int ei = 8 * h + v;
    if (e0 + ei < e_pin) {
      const _Float16* hvrow = hv + (size_t)sS[wave][ei] * 64;
      float* arow = acc + (size_t)sD[wave][ei] * 64;
#pragma unroll
      for (int nt = 0; nt < 4; ++nt) {
        int col = nt * 16 + mn;
        atomicAdd(arow + col, he[nt][v] * (float)hvrow[col]);
      }
    }
  }
}

// ---------------------------------------------------------------------------
// K3: cell_out = ssp(acc @ W_cfo + b_cfo); collapse readout weights to 4
//     per-cell scalars: pd1/pd2 (dis halves), pa1/pa2 (ang halves)
// ---------------------------------------------------------------------------
__global__ __launch_bounds__(128) void k_cellout(
    const float* __restrict__ acc,
    const float* __restrict__ W_cfo, const float* __restrict__ b_cfo,
    const float* __restrict__ W_dis, const float* __restrict__ W_ang,
    float* __restrict__ pd1, float* __restrict__ pd2,
    float* __restrict__ pa1, float* __restrict__ pa2, int n_cell) {
  __shared__ _Float16 sWT[64 * 64];    // transposed [n][k]
  __shared__ float sB[64];
  __shared__ float sWd[4][64];

  const int tid = threadIdx.x;
  for (int i = tid; i < 64 * 64; i += 128) {
    int n = i >> 6, k = i & 63;
    sWT[i] = (_Float16)W_cfo[k * 64 + n];
  }
  if (tid < 64) {
    sB[tid] = b_cfo[tid];
    sWd[0][tid] = W_dis[tid];
    sWd[1][tid] = W_dis[64 + tid];
    sWd[2][tid] = W_ang[tid];
    sWd[3][tid] = W_ang[64 + tid];
  }
  __syncthreads();

  const int wave = tid >> 5, lane = tid & 31;
  const int mn = lane & 15, h = lane >> 4;
  const int tile = blockIdx.x * 4 + wave;
  const int row0 = tile * 16;
  if (row0 >= n_cell) return;

  int rA = row0 + mn; rA = rA < n_cell ? rA : n_cell - 1;
  const float* grow = acc + (size_t)rA * 64;
  v16h a0 = gA8(grow, h, 0);
  v16h a1 = gA8(grow, h, 32);

  float co[4][8];
#pragma unroll
  for (int nt = 0; nt < 4; ++nt) {
    v8f c = wmma16(a0, ldsBT(sWT, 64, nt * 16 + mn, h, 0), v8zero());
    c = wmma16(a1, ldsBT(sWT, 64, nt * 16 + mn, h, 32), c);
    float bb = sB[nt * 16 + mn];
#pragma unroll
    for (int v = 0; v < 8; ++v) co[nt][v] = ssp_fast(c[v] + bb);
  }

  float* outs[4] = {pd1, pd2, pa1, pa2};
#pragma unroll
  for (int w = 0; w < 4; ++w) {
#pragma unroll
    for (int v = 0; v < 8; ++v) {
      float p = co[0][v] * sWd[w][mn] + co[1][v] * sWd[w][16 + mn]
              + co[2][v] * sWd[w][32 + mn] + co[3][v] * sWd[w][48 + mn];
      p += __shfl_xor(p, 1);
      p += __shfl_xor(p, 2);
      p += __shfl_xor(p, 4);
      p += __shfl_xor(p, 8);   // stays within the 16-lane half (row owner)
      int r = row0 + 8 * h + v;
      if (mn == 0 && r < n_cell) outs[w][r] = p;
    }
  }
}

// ---------------------------------------------------------------------------
// K4: edge readout — scalar gathers only
// ---------------------------------------------------------------------------
__global__ void k_edges(const int* __restrict__ fathers, const int* __restrict__ sons,
                        const float* __restrict__ pd1, const float* __restrict__ pd2,
                        const float* __restrict__ pa1, const float* __restrict__ pa2,
                        const float* __restrict__ b_dis, const float* __restrict__ b_ang,
                        float* __restrict__ out, int e_pt) {
  int e = blockIdx.x * blockDim.x + threadIdx.x;
  if (e < e_pt) {
    int f = fathers[e], s = sons[e];
    out[e] = __expf(pd1[f] + pd2[s] + b_dis[0]);
    out[e_pt + e] = tanh_fast(pa1[f] + pa2[s] + b_ang[0]) * 4.0f;
  }
}

extern "C" void kernel_launch(void* const* d_in, const int* in_sizes, int n_in,
                              void* d_out, int out_size, void* d_ws, size_t ws_size,
                              hipStream_t stream) {
  const float* net_feat  = (const float*)d_in[1];
  const float* pin_feat  = (const float*)d_in[2];
  const int* pinned_src  = (const int*)d_in[5];
  const int* pinned_dst  = (const int*)d_in[6];
  const int* fathers     = (const int*)d_in[7];
  const int* sons        = (const int*)d_in[8];
  const float* W_net  = (const float*)d_in[11];
  const float* b_net  = (const float*)d_in[12];
  const float* W_pin  = (const float*)d_in[13];
  const float* b_pin  = (const float*)d_in[14];
  const float* W_cfn  = (const float*)d_in[17];
  const float* b_cfn  = (const float*)d_in[18];
  const float* W_cfe1 = (const float*)d_in[19];
  const float* b_cfe1 = (const float*)d_in[20];
  const float* W_cfe2 = (const float*)d_in[21];
  const float* b_cfe2 = (const float*)d_in[22];
  const float* W_cfo  = (const float*)d_in[23];
  const float* b_cfo  = (const float*)d_in[24];
  const float* W_dis  = (const float*)d_in[25];
  const float* b_dis  = (const float*)d_in[26];
  const float* W_ang  = (const float*)d_in[27];
  const float* b_ang  = (const float*)d_in[28];

  const int N_CELL = in_sizes[0] / 8;
  const int N_NET  = in_sizes[1] / 4;
  const int E_PIN  = in_sizes[2] / 4;
  const int E_PT   = in_sizes[7];

  char* ws = (char*)d_ws;
  auto alignup = [](size_t x) { return (x + 255) & ~(size_t)255; };
  size_t off = 0;
  _Float16* hv = (_Float16*)(ws + off); off = alignup(off + (size_t)N_NET * 64 * 2);
  float* accb  = (float*)(ws + off);    off = alignup(off + (size_t)N_CELL * 64 * 4);
  float* pd1   = (float*)(ws + off);    off = alignup(off + (size_t)N_CELL * 4);
  float* pd2   = (float*)(ws + off);    off = alignup(off + (size_t)N_CELL * 4);
  float* pa1   = (float*)(ws + off);    off = alignup(off + (size_t)N_CELL * 4);
  float* pa2   = (float*)(ws + off);    off = alignup(off + (size_t)N_CELL * 4);

  hipMemsetAsync(accb, 0, (size_t)N_CELL * 64 * sizeof(float), stream);

  int tilesNet = (N_NET + 15) / 16;
  k_hv<<<(tilesNet + 3) / 4, 128, 0, stream>>>(net_feat, W_net, b_net, W_cfn, b_cfn,
                                               hv, N_NET);

  int tilesPin = (E_PIN + 15) / 16;
  k_cfconv<<<(tilesPin + 3) / 4, 128, 0, stream>>>(pin_feat, pinned_src, pinned_dst,
      W_pin, b_pin, W_cfe1, b_cfe1, W_cfe2, b_cfe2, hv, accb, E_PIN);

  int tilesCell = (N_CELL + 15) / 16;
  k_cellout<<<(tilesCell + 3) / 4, 128, 0, stream>>>(accb, W_cfo, b_cfo, W_dis, W_ang,
      pd1, pd2, pa1, pa2, N_CELL);

  k_edges<<<(E_PT + 255) / 256, 256, 0, stream>>>(fathers, sons, pd1, pd2, pa1, pa2,
      b_dis, b_ang, (float*)d_out, E_PT);
}
